// PiecewiseRigidDecoder_39565238730808
// MI455X (gfx1250) — matrix operand order repaired and checked by reading, not compiled, not run on hardware
//
#include <hip/hip_runtime.h>

// PiecewiseRigidDecoder on MI455X (gfx1250, wave32, WMMA).
//
// Roofline: 103 GFLOP of 256x256 matmuls vs ~30MB unique bytes -> matrix-core
// bound. Path: f16 WMMA (v_wmma_f32_16x16x32_f16) with f32 accumulation.
// Weights are converted f32->f16 and pre-swizzled into WMMA-B register layout
// ONCE by a prep kernel into d_ws (6MB); the main kernel streams them into LDS
// with async global->LDS copies (ASYNCcnt / s_wait_asynccnt). Activations live
// in WMMA-C register layout (v8f per 16x16 tile); LayerNorm uses wave32
// shfl_xor reductions; an f16 activation image in LDS feeds the A operand.
// B tiles are double-buffered in registers (prefetch distance 2) so ds_load
// latency overlaps the WMMA chain instead of serializing on s_wait_dscnt 0.

typedef __attribute__((ext_vector_type(16))) _Float16 v16h;
typedef __attribute__((ext_vector_type(8)))  _Float16 v8h;
typedef __attribute__((ext_vector_type(8)))  float    v8f;
typedef __attribute__((ext_vector_type(4)))  float    v4f;

// Async global->LDS builtin takes (global int4*, shared int4*, imm, imm).
typedef int v4i_vs __attribute__((vector_size(16)));
typedef __attribute__((address_space(1))) v4i_vs* async_gptr;
typedef __attribute__((address_space(3))) v4i_vs* async_lptr;

#define HH     256
#define SLOPE  0.1f
#define EPSLN  1e-5f

#if __has_builtin(__builtin_amdgcn_global_load_async_to_lds_b128)
#define HAVE_ASYNC_LDS 1
#endif

static __device__ inline void wait_async0() {
#if __has_builtin(__builtin_amdgcn_s_wait_asynccnt)
  __builtin_amdgcn_s_wait_asynccnt(0);
#else
  asm volatile("s_wait_asynccnt 0" ::: "memory");
#endif
}

static __device__ inline v16h cat16(v8h a, v8h b) {
  return __builtin_shufflevector(a, b, 0,1,2,3,4,5,6,7,8,9,10,11,12,13,14,15);
}

// B tile (ot,kt) from the B-layout LDS image: 32B contiguous per lane.
static __device__ inline v16h loadB(const _Float16* sW, int ot, int kt, int lane) {
  const v8h* bp = (const v8h*)(sW + (ot * 8 + kt) * 512 + lane * 16);
  return cat16(bp[0], bp[1]);
}

// LayerNorm over 256 cols + LeakyReLU, operating on C-layout tiles.
// Lane L holds rows {e + 8*(L/16)} at cols {16t + L%16}; row reductions are
// partial sums over t reduced across the 16-lane group via shfl_xor.
static __device__ inline void ln_lrelu(v8f (&v)[16], const float* sP,
                                       int goff, int beoff, int laneLo)
{
  float mu[8], rs[8];
  #pragma unroll
  for (int e = 0; e < 8; ++e) {
    float s = 0.f, ss = 0.f;
    #pragma unroll
    for (int t = 0; t < 16; ++t) { float a = v[t][e]; s += a; ss += a * a; }
    #pragma unroll
    for (int m = 1; m < 16; m <<= 1) {
      s  += __shfl_xor(s,  m, 32);
      ss += __shfl_xor(ss, m, 32);
    }
    float m1  = s  * (1.f / 256.f);
    float var = ss * (1.f / 256.f) - m1 * m1;
    mu[e] = m1;
    rs[e] = rsqrtf(var + EPSLN);
  }
  #pragma unroll
  for (int t = 0; t < 16; ++t) {
    int col  = t * 16 + laneLo;
    float g  = sP[goff  + col];
    float be = sP[beoff + col];
    #pragma unroll
    for (int e = 0; e < 8; ++e) {
      float a = (v[t][e] - mu[e]) * rs[e] * g + be;
      v[t][e] = a >= 0.f ? a : SLOPE * a;
    }
  }
}

// B-register-layout swizzle for tile(ot,kt): lane = o%16 + 16*((k%32)/16),
// half index = k%16.  Linear image: ((ot*8+kt)*32 + lane)*16 + half.
static __device__ inline int bswizzle(int k, int o) {
  int kt = k >> 5, kk = k & 31;
  int ot = o >> 4;
  int ln = (o & 15) | ((kk >> 4) << 4);
  return (ot * 8 + kt) * 512 + ln * 16 + (kk & 15);
}

// ---- prep: convert Wm (3,16,256,256) f32 -> f16 pre-swizzled B-layout ----
extern "C" __global__ __launch_bounds__(256) void prep_weights(
    const float* __restrict__ Wm, _Float16* __restrict__ wsW)
{
  const int mat = blockIdx.x;                 // l*16+n, 0..47
  const int tid = threadIdx.x;
  const v4f* Wg = (const v4f*)(Wm + ((size_t)mat << 16));
  _Float16* dst = wsW + ((size_t)mat << 16);
  for (int it = 0; it < 64; ++it) {
    int f  = it * 256 + tid;                  // float4 index, o fastest
    v4f w4 = Wg[f];
    int k  = f >> 6;
    int ob = (f & 63) << 2;
    #pragma unroll
    for (int j = 0; j < 4; ++j)
      dst[bswizzle(k, ob + j)] = (_Float16)w4[j];
  }
}

// ---- main kernel; PRE: weights pre-swizzled f16 in wsW ----
template <bool PRE>
__global__ __launch_bounds__(256) void prd_kernel(
    const float* __restrict__ xyz,  const float* __restrict__ latent,
    const float* __restrict__ W0,   const float* __restrict__ b0,
    const float* __restrict__ g0,   const float* __restrict__ be0,
    const float* __restrict__ Wm,   const float* __restrict__ bm,
    const float* __restrict__ gm,   const float* __restrict__ bem,
    const float* __restrict__ Wf,   const float* __restrict__ bf,
    const _Float16* __restrict__ wsW, float* __restrict__ out)
{
  __shared__ _Float16 sW[HH * HH];       // 128KB: one layer's weights, B-layout
  __shared__ _Float16 sX[8 * 16 * HH];   //  64KB: per-wave 16x256 f16 activations
  __shared__ float    sP[4096];          //  16KB: biases / gamma / beta / Wf

  const int tid    = threadIdx.x;
  const int lane   = tid & 31;
  const int wave   = tid >> 5;
  const int laneLo = lane & 15;
  const int laneHi = lane >> 4;
  const int n      = blockIdx.x & 15;        // bone
  const int blk    = blockIdx.x >> 4;        // row block (128 rows)
  const int rowbase = blk * 128 + wave * 16; // this wave's first batch row

  // ---- stage per-bone small params into LDS (blockDim.x == 256) ----
  for (int i = tid; i < 768; i += 256) sP[i] = W0[n * 768 + i];     // W0[n][i][o]
  sP[ 768 + tid] = b0 [n * 256 + tid];
  sP[1024 + tid] = g0 [n * 256 + tid];
  sP[1280 + tid] = be0[n * 256 + tid];
  #pragma unroll
  for (int l = 0; l < 3; ++l) {
    sP[1536 + l * 768       + tid] = bm [(l * 16 + n) * 256 + tid];
    sP[1536 + l * 768 + 256 + tid] = gm [(l * 16 + n) * 256 + tid];
    sP[1536 + l * 768 + 512 + tid] = bem[(l * 16 + n) * 256 + tid];
  }
  sP[3840 + tid] = Wf[n * 256 + tid];
  __syncthreads();

  // ---- rigid transform: inp[b][n][:3] = latent[b][n] @ [xyz,1] ----
  v8f x[16];
  {
    int bi = rowbase + laneLo;  // lanes 16-31 redo rows 0..15 (harmless)
    float hx = xyz[bi * 3 + 0], hy = xyz[bi * 3 + 1], hz = xyz[bi * 3 + 2];
    const v4f* lat = (const v4f*)(latent + (size_t)(bi * 16 + n) * 16);
    v4f l0 = lat[0], l1 = lat[1], l2 = lat[2];
    float i0 = l0[0] * hx + l0[1] * hy + l0[2] * hz + l0[3];
    float i1 = l1[0] * hx + l1[1] * hy + l1[2] * hz + l1[3];
    float i2 = l2[0] * hx + l2[1] * hy + l2[2] * hz + l2[3];
    float r0[8], r1[8], r2[8];
    #pragma unroll
    for (int e = 0; e < 8; ++e) {
      int src = e + 8 * laneHi;              // broadcast row e(+8)'s inp
      r0[e] = __shfl(i0, src, 32);
      r1[e] = __shfl(i1, src, 32);
      r2[e] = __shfl(i2, src, 32);
    }
    // layer 0: K=3 linear in VALU, result directly in C layout
    #pragma unroll
    for (int t = 0; t < 16; ++t) {
      int col = t * 16 + laneLo;
      float wa = sP[col], wb = sP[256 + col], wc = sP[512 + col], bb = sP[768 + col];
      #pragma unroll
      for (int e = 0; e < 8; ++e)
        x[t][e] = r0[e] * wa + r1[e] * wb + r2[e] * wc + bb;
    }
  }
  ln_lrelu(x, sP, 1024, 1280, laneLo);

  _Float16* xw = sX + wave * 4096;  // this wave's 16x256 f16 activation image
  #pragma unroll
  for (int t = 0; t < 16; ++t)
    #pragma unroll
    for (int e = 0; e < 8; ++e)
      xw[(e + 8 * laneHi) * 256 + t * 16 + laneLo] = (_Float16)x[t][e];

  // ---- 3 middle layers: Linear(256x256) -> LN -> LeakyReLU -> +residual ----
  for (int l = 0; l < 3; ++l) {
    __syncthreads();  // previous layer's sW reads complete
    if (PRE) {
      // Pre-swizzled f16 image in d_ws: pure linear 128KB copy into LDS.
      const v8h* Wg = (const v8h*)(wsW + ((size_t)(l * 16 + n) << 16));
      v8h* sWv = (v8h*)sW;
      for (int it = 0; it < 32; ++it) {
        int idx = it * 256 + tid;        // 16B chunk, coalesced
#ifdef HAVE_ASYNC_LDS
        __builtin_amdgcn_global_load_async_to_lds_b128(
            (async_gptr)(Wg + idx), (async_lptr)(sWv + idx), 0, 0);
#else
        sWv[idx] = Wg[idx];
#endif
      }
#ifdef HAVE_ASYNC_LDS
      wait_async0();
#endif
    } else {
      // Fallback: convert+swizzle Wm[l][n] (f32 row-major [k][o]) in-kernel.
      const v4f* Wg = (const v4f*)(Wm + ((size_t)(l * 16 + n) << 16));
      for (int it = 0; it < 64; ++it) {
        int f  = it * 256 + tid;
        v4f w4 = Wg[f];
        int k  = f >> 6;
        int ob = (f & 63) << 2;
        #pragma unroll
        for (int j = 0; j < 4; ++j)
          sW[bswizzle(k, ob + j)] = (_Float16)w4[j];
      }
    }
    __syncthreads();

    v8f acc[16];
    v8f zero = {0.f, 0.f, 0.f, 0.f, 0.f, 0.f, 0.f, 0.f};
    #pragma unroll
    for (int t = 0; t < 16; ++t) acc[t] = zero;

    #pragma unroll
    for (int kt = 0; kt < 8; ++kt) {
      // A 16x32 f16: lane=row, halves = {16*laneHi+0..15}
      const v8h* ap = (const v8h*)(xw + laneLo * 256 + kt * 32 + laneHi * 16);
      v16h A = cat16(ap[0], ap[1]);
      // Double-buffered B tiles: prefetch distance 2 keeps 4 ds_loads in
      // flight so the WMMA chain waits at dscnt<=4 instead of 0.
      v16h bCur = loadB(sW, 0, kt, lane);
      v16h bNxt = loadB(sW, 1, kt, lane);
      #pragma unroll
      for (int ot = 0; ot < 16; ++ot) {
        v16h bPre = (ot < 14) ? loadB(sW, ot + 2, kt, lane) : bNxt;
        acc[ot] = __builtin_amdgcn_wmma_f32_16x16x32_f16(
            false, A, false, bCur, (short)0, acc[ot], false, false);
        bCur = bNxt;
        bNxt = bPre;
      }
    }

    int pb = 1536 + l * 768;
    #pragma unroll
    for (int t = 0; t < 16; ++t) {
      float bb = sP[pb + t * 16 + laneLo];
      #pragma unroll
      for (int e = 0; e < 8; ++e) acc[t][e] += bb;
    }
    ln_lrelu(acc, sP, pb + 256, pb + 512, laneLo);
    #pragma unroll
    for (int t = 0; t < 16; ++t) x[t] += acc[t];  // residual (f32)

    #pragma unroll
    for (int t = 0; t < 16; ++t)
      #pragma unroll
      for (int e = 0; e < 8; ++e)
        xw[(e + 8 * laneHi) * 256 + t * 16 + laneLo] = (_Float16)x[t][e];
  }

  // ---- final: out[b][n] = x . Wf[n] + bf[n] ----
  float bfv = bf[n];
  float p[8];
  #pragma unroll
  for (int e = 0; e < 8; ++e) {
    float s = 0.f;
    #pragma unroll
    for (int t = 0; t < 16; ++t) s += x[t][e] * sP[3840 + t * 16 + laneLo];
    #pragma unroll
    for (int m = 1; m < 16; m <<= 1) s += __shfl_xor(s, m, 32);
    p[e] = s;
  }
  if (laneLo == 0) {
    #pragma unroll
    for (int e = 0; e < 8; ++e) {
      int b2 = rowbase + e + 8 * laneHi;
      out[b2 * 16 + n] = p[e] + bfv;
    }
  }
}

extern "C" void kernel_launch(void* const* d_in, const int* in_sizes, int n_in,
                              void* d_out, int out_size, void* d_ws, size_t ws_size,
                              hipStream_t stream) {
  (void)in_sizes; (void)n_in; (void)out_size;
  const float* xyz    = (const float*)d_in[0];
  const float* latent = (const float*)d_in[1];
  const float* W0     = (const float*)d_in[2];
  const float* b0     = (const float*)d_in[3];
  const float* g0     = (const float*)d_in[4];
  const float* be0    = (const float*)d_in[5];
  const float* Wm     = (const float*)d_in[6];
  const float* bm     = (const float*)d_in[7];
  const float* gm     = (const float*)d_in[8];
  const float* bem    = (const float*)d_in[9];
  const float* Wf     = (const float*)d_in[10];
  const float* bf     = (const float*)d_in[11];

  const size_t need = (size_t)3 * 16 * 256 * 256 * sizeof(_Float16);  // 6MB
  if (ws_size >= need) {
    _Float16* wsW = (_Float16*)d_ws;
    hipLaunchKernelGGL(prep_weights, dim3(48), dim3(256), 0, stream, Wm, wsW);
    hipLaunchKernelGGL(HIP_KERNEL_NAME(prd_kernel<true>), dim3(2048), dim3(256),
                       0, stream, xyz, latent, W0, b0, g0, be0, Wm, bm, gm, bem,
                       Wf, bf, (const _Float16*)wsW, (float*)d_out);
  } else {
    hipLaunchKernelGGL(HIP_KERNEL_NAME(prd_kernel<false>), dim3(2048), dim3(256),
                       0, stream, xyz, latent, W0, b0, g0, be0, Wm, bm, gm, bem,
                       Wf, bf, (const _Float16*)nullptr, (float*)d_out);
  }
}
